// KGCN_E_15126874816996
// MI455X (gfx1250) — matrix operand order repaired and checked by reading, not compiled, not run on hardware
//
#include <hip/hip_runtime.h>
#include <hip/hip_bf16.h>

typedef __attribute__((ext_vector_type(16))) _Float16 v16h;
typedef __attribute__((ext_vector_type(8)))  _Float16 v8h;
typedef __attribute__((ext_vector_type(8)))  float    v8f;

#define DIMK   32   // embedding dim
#define KNB    16   // neighbors per node
#define BTILE  16   // batch elements per block
#define THREADS 256 // 8 wave32s

// ---- LDS layout offsets (bytes), total 58496 < 64KB ----
#define OFF_USER   0        // float[16*32]  user vectors         2048
#define OFF_BIAS   2048     // float[32]     bias                  128
#define OFF_SCORE  2176     // float[16*32]  user.rel score table 2048
#define OFF_EV1    4224     // f16[16*16*32] hop-1 embeddings    16384
#define OFF_E1P    20608    // f16[16*16*32] hop-1 updated       16384
#define OFF_X1     36992    // f16[16*16*32] hop-1 matmul input  16384
#define OFF_X0     53376    // f16[16*32]    hop-0 matmul input   1024
#define OFF_E0     54400    // float[16*32]  hop-0 result         2048
#define OFF_ATT0   56448    // float[16*16]  hop-0 attention      1024
#define OFF_IDX1   57472    // int[256]      hop-1 entity ids     1024
#define SMEM_BYTES 58496

__device__ __forceinline__ float sigmoidf_(float x) {
  return __builtin_amdgcn_rcpf(1.0f + __expf(-x));   // v_rcp_f32, no div expansion
}

__device__ __forceinline__ float tanhf_(float x) {
#if defined(__gfx1250__) && __has_builtin(__builtin_amdgcn_tanhf)
  return __builtin_amdgcn_tanhf(x);                  // gfx1250 V_TANH_F32
#else
  float t = __expf(2.0f * x);
  return (t - 1.0f) * __builtin_amdgcn_rcpf(t + 1.0f);
#endif
}

// Build the two B fragments (columns 0-15 / 16-31 of Mt^T) for a 32x32
// row-major f32 matrix M, per the CDNA5 16-bit B layout:
// lane%16 = N, halves 0..7 = K base..base+7, halves 8..15 = K base+16.., base = (lane<16 ? 0 : 8). B[k][n] = M[n][k].
__device__ __forceinline__ void load_bt(const float* __restrict__ M,
                                        int col, int hiH, v16h bt[2]) {
#pragma unroll
  for (int t = 0; t < 2; ++t)
#pragma unroll
    for (int h = 0; h < 16; ++h)
      bt[t][h] = (_Float16)M[(col + 16 * t) * DIMK + ((h < 8 ? h : h + 8) + hiH * 8)];
}

// One [16 x 32] @ W^T [32 x 32] + bias, activation, scatter D to LDS.
// Xrow: f16 row-major [16][32] in LDS. A fragment per the 16-bit A layout:
// lane%16 = M; halves 0..7 = K base..base+7, halves 8..15 = K base+16..,
// base = (lane<16 ? 0 : 8).
template <typename OutT>
__device__ __forceinline__ void wmma_rowblock(const _Float16* Xrow, const v16h btile[2],
                                              const float biasv[2], OutT* outLds,
                                              int col, int hiH, int act) {
  const char* bp = (const char*)Xrow + col * 64 + hiH * 16;
  v8h lo = *(const v8h*)(bp);
  v8h hi = *(const v8h*)(bp + 32);
  v16h a;
#pragma unroll
  for (int h = 0; h < 8; ++h) { a[h] = lo[h]; a[8 + h] = hi[h]; }
#pragma unroll
  for (int t = 0; t < 2; ++t) {
    v8f c;
#pragma unroll
    for (int j = 0; j < 8; ++j) c[j] = biasv[t];
    v8f d = __builtin_amdgcn_wmma_f32_16x16x32_f16(false, a, false, btile[t],
                                                   (short)0, c, false, false);
#pragma unroll
    for (int j = 0; j < 8; ++j) {
      float val = d[j];
      val = act ? tanhf_(val) : sigmoidf_(val);
      // D layout: lane holds out[m = j + (lane<16?0:8)][n = lane%16 + 16t]
      outLds[(j + hiH * 8) * DIMK + (col + 16 * t)] = (OutT)val;
    }
  }
}

__global__ __launch_bounds__(THREADS)
void kgcn_kernel(const int* __restrict__ u, const int* __restrict__ v,
                 const float* __restrict__ usr_w, const float* __restrict__ ent_w,
                 const float* __restrict__ rel_w, const float* __restrict__ W,
                 const float* __restrict__ bias,
                 const int* __restrict__ adj_ent, const int* __restrict__ adj_rel,
                 float* __restrict__ out, int Btot) {
  extern __shared__ char smem[];
  float*    userLds  = (float*)(smem + OFF_USER);
  float*    biasLds  = (float*)(smem + OFF_BIAS);
  float*    scoreTab = (float*)(smem + OFF_SCORE);   // [16 b][32 r]
  _Float16* ev1Lds   = (_Float16*)(smem + OFF_EV1);
  _Float16* e1pLds   = (_Float16*)(smem + OFF_E1P);
  _Float16* X1Lds    = (_Float16*)(smem + OFF_X1);
  _Float16* X0Lds    = (_Float16*)(smem + OFF_X0);
  float*    e0Lds    = (float*)(smem + OFF_E0);
  float*    att0Lds  = (float*)(smem + OFF_ATT0);
  int*      idx1Lds  = (int*)(smem + OFF_IDX1);

  const int tid   = threadIdx.x;
  const int lane  = tid & 31;
  const int wave  = tid >> 5;
  const int col   = lane & 15;   // N index / M row for A fragment
  const int hiH   = lane >> 4;   // 0: lanes 0-15, 1: lanes 16-31
  const int bBase = blockIdx.x * BTILE;

  // ---- Phase 0: stage users / bias / hop-1 indices ----
  if (tid < DIMK) biasLds[tid] = bias[tid];
  for (int e = tid; e < BTILE * DIMK; e += THREADS) {
    int b = e >> 5, d = e & 31;
    int bg = bBase + b; if (bg >= Btot) bg = Btot - 1;
    userLds[e] = usr_w[u[bg] * DIMK + d];
  }
  {
    int b = tid >> 4, n = tid & 15;
    int bg = bBase + b; if (bg >= Btot) bg = Btot - 1;
    idx1Lds[tid] = adj_ent[v[bg] * KNB + n];
  }
  __syncthreads();

  // ---- Per-wave B fragments for W^T + bias ----
  v16h wbt[2];
  load_bt(W, col, hiH, wbt);
  float biasv[2] = { biasLds[col], biasLds[col + 16] };

  // ---- Phase 1: score table S[b][r] = user[b] . rel_w[r] via WMMA (wave 0) ----
  if (wave == 0) {
    v16h rbt[2];
    load_bt(rel_w, col, hiH, rbt);
    v16h a;
#pragma unroll
    for (int h = 0; h < 16; ++h) {
      int K = (h < 8 ? h : h + 8) + hiH * 8;
      a[h] = (_Float16)userLds[col * DIMK + K];
    }
#pragma unroll
    for (int t = 0; t < 2; ++t) {
      v8f c = {};
      v8f d = __builtin_amdgcn_wmma_f32_16x16x32_f16(false, a, false, rbt[t],
                                                     (short)0, c, false, false);
#pragma unroll
      for (int j = 0; j < 8; ++j)
        scoreTab[(j + hiH * 8) * DIMK + (col + 16 * t)] = d[j];
    }
  }
  __syncthreads();

  // ---- Phase 2: iter0/hop1 attention + aggregation (wave per node, lane=dim) ----
#pragma unroll 1
  for (int i = 0; i < 32; ++i) {
    int node = wave * 32 + i;           // 256 nodes = 16 b x 16 n
    int b = node >> 4;
    int e1 = idx1Lds[node];
    int myidx = (lane < 16) ? adj_ent[e1 * KNB + lane]
                            : adj_rel[e1 * KNB + (lane - 16)];
    // lanes 16-31 hold score for neighbor k = lane-16 (lanes 0-15: garbage, unused)
    float sc = scoreTab[b * DIMK + (myidx & 31)];
    float mx = sc;
    mx = fmaxf(mx, __shfl_xor(mx, 8, 32));
    mx = fmaxf(mx, __shfl_xor(mx, 4, 32));
    mx = fmaxf(mx, __shfl_xor(mx, 2, 32));
    mx = fmaxf(mx, __shfl_xor(mx, 1, 32));
    float ex = __expf(sc - mx);
    float den = ex;
    den += __shfl_xor(den, 8, 32);
    den += __shfl_xor(den, 4, 32);
    den += __shfl_xor(den, 2, 32);
    den += __shfl_xor(den, 1, 32);
    float attn = ex * __builtin_amdgcn_rcpf(den);
    float agg = 0.0f;
#pragma unroll
    for (int k = 0; k < 16; ++k) {
      float ak = __shfl(attn, 16 + k, 32);
      int e = __shfl(myidx, k, 32);
      agg += ak * ent_w[e * DIMK + lane];   // coalesced 128B row gather
    }
    float ev1 = ent_w[e1 * DIMK + lane];
    ev1Lds[node * DIMK + lane] = (_Float16)ev1;
    X1Lds[node * DIMK + lane]  = (_Float16)(ev1 + agg);
  }
  __syncthreads();

  // ---- Phase 3: iter0/hop1 Linear via WMMA (wave handles 2 batch elems) ----
#pragma unroll
  for (int bb = 0; bb < 2; ++bb) {
    int b = wave * 2 + bb;
    wmma_rowblock<_Float16>(X1Lds + b * (16 * DIMK), wbt, biasv,
                            e1pLds + b * (16 * DIMK), col, hiH, /*sigmoid*/0);
  }
  __syncthreads();

  // ---- Phase 4: iter0/hop0 attention + aggregation (wave per 2 batch elems) ----
#pragma unroll
  for (int bb = 0; bb < 2; ++bb) {
    int b = wave * 2 + bb;
    int bg = bBase + b; if (bg >= Btot) bg = Btot - 1;
    int vb = v[bg];
    int ridx = adj_rel[vb * KNB + col];   // both halves load identical values
    float sc = scoreTab[b * DIMK + (ridx & 31)];
    float mx = sc;
    mx = fmaxf(mx, __shfl_xor(mx, 8, 32));
    mx = fmaxf(mx, __shfl_xor(mx, 4, 32));
    mx = fmaxf(mx, __shfl_xor(mx, 2, 32));
    mx = fmaxf(mx, __shfl_xor(mx, 1, 32));
    float ex = __expf(sc - mx);
    float den = ex;
    den += __shfl_xor(den, 8, 32);
    den += __shfl_xor(den, 4, 32);
    den += __shfl_xor(den, 2, 32);
    den += __shfl_xor(den, 1, 32);
    float attn = ex * __builtin_amdgcn_rcpf(den);
    if (lane < 16) att0Lds[b * 16 + lane] = attn;   // reused by iter1/hop0
    float agg = 0.0f;
#pragma unroll
    for (int k = 0; k < 16; ++k) {
      float ak = __shfl(attn, k, 32);
      agg += ak * (float)ev1Lds[(b * 16 + k) * DIMK + lane];
    }
    float x = ent_w[vb * DIMK + lane] + agg;
    X0Lds[b * DIMK + lane] = (_Float16)x;
  }
  __syncthreads();

  // ---- Phase 5a: iter0/hop0 Linear (16 batch rows stacked -> one WMMA pair) ----
  if (wave == 0)
    wmma_rowblock<float>(X0Lds, wbt, biasv, e0Lds, col, hiH, /*sigmoid*/0);
  __syncthreads();

  // ---- Phase 5b: iter1/hop0 aggregation (attention identical to iter0/hop0) ----
#pragma unroll
  for (int bb = 0; bb < 2; ++bb) {
    int b = wave * 2 + bb;
    float agg = 0.0f;
#pragma unroll
    for (int k = 0; k < 16; ++k)
      agg += att0Lds[b * 16 + k] * (float)e1pLds[(b * 16 + k) * DIMK + lane];
    X0Lds[b * DIMK + lane] = (_Float16)(e0Lds[b * DIMK + lane] + agg);
  }
  __syncthreads();

  // ---- Phase 5c: iter1/hop0 Linear + tanh ----
  if (wave == 0)
    wmma_rowblock<float>(X0Lds, wbt, biasv, e0Lds, col, hiH, /*tanh*/1);
  __syncthreads();

  // ---- Phase 6: final user . item, sigmoid ----
  if (tid < BTILE) {
    int bg = bBase + tid;
    if (bg < Btot) {
      float s = 0.0f;
#pragma unroll
      for (int d = 0; d < DIMK; ++d)
        s += userLds[tid * DIMK + d] * e0Lds[tid * DIMK + d];
      out[bg] = sigmoidf_(s);
    }
  }
}

extern "C" void kernel_launch(void* const* d_in, const int* in_sizes, int n_in,
                              void* d_out, int out_size, void* d_ws, size_t ws_size,
                              hipStream_t stream) {
  (void)n_in; (void)d_ws; (void)ws_size; (void)out_size;
  const int*   u       = (const int*)d_in[0];
  const int*   v       = (const int*)d_in[1];
  const float* usr_w   = (const float*)d_in[2];
  const float* ent_w   = (const float*)d_in[3];
  const float* rel_w   = (const float*)d_in[4];
  const float* W       = (const float*)d_in[5];
  const float* bias    = (const float*)d_in[6];
  const int*   adj_ent = (const int*)d_in[7];
  const int*   adj_rel = (const int*)d_in[8];
  float*       out     = (float*)d_out;

  int B = in_sizes[0];
  int grid = (B + BTILE - 1) / BTILE;
  kgcn_kernel<<<grid, THREADS, SMEM_BYTES, stream>>>(
      u, v, usr_w, ent_w, rel_w, W, bias, adj_ent, adj_rel, out, B);
}
